// FrameTransformerDecoder_6622839570820
// MI455X (gfx1250) — matrix operand order repaired and checked by reading, not compiled, not run on hardware
//
#include <hip/hip_runtime.h>
#include <hip/hip_bf16.h>

// ---------------------------------------------------------------------------
// FrameTransformerDecoder for MI455X (gfx1250, wave32, WMMA)
// Heavy GEMMs (mc_linear, QK, AV, MLP) -> v_wmma_f32_16x16x32_f16
// GEMM staging: W-slab via TENSOR_LOAD_TO_LDS (TDM, TENSORcnt),
//               X-slab via GLOBAL_LOAD_ASYNC_TO_LDS_B128 (ASYNCcnt)
// ---------------------------------------------------------------------------

typedef _Float16 v16h __attribute__((ext_vector_type(16)));
typedef float    v8f  __attribute__((ext_vector_type(8)));
typedef unsigned int u32x4 __attribute__((ext_vector_type(4)));
typedef int          i32x4 __attribute__((ext_vector_type(4)));
typedef int          i32x8 __attribute__((ext_vector_type(8)));

#define BD   2
#define CD   8
#define FD   512
#define WD   256
#define HD_  64
#define HEADS_ 8
#define EXP_ 2048

static constexpr size_t FSZ  = (size_t)BD * CD * FD * WD;                 // 2,097,152
static constexpr size_t QKSZ = (size_t)BD * CD * HEADS_ * WD * WD;        // 8,388,608
static constexpr size_t GSZ  = (size_t)BD * HEADS_ * WD * WD;             // 1,048,576
static constexpr size_t CATSZ = (size_t)BD * 24 * FD * WD;                // 6,291,456

// -------------------------- WMMA / async / TDM helpers ---------------------

__device__ __forceinline__ v8f wmma_f16(v16h a, v16h b, v8f c) {
  return __builtin_amdgcn_wmma_f32_16x16x32_f16(false, a, false, b, (short)0, c,
                                                false, false);
}

// Async copy 16 bytes global -> LDS (per-lane addresses). Tracked by ASYNCcnt.
__device__ __forceinline__ void async_g2l_b128(unsigned lds_byte_off,
                                               const void* gsrc) {
  asm volatile("global_load_async_to_lds_b128 %0, %1, off"
               :
               : "v"(lds_byte_off), "v"(gsrc)
               : "memory");
}

__device__ __forceinline__ void wait_async0() {
  asm volatile("s_wait_asynccnt 0x0" ::: "memory");
}

#if __has_builtin(__builtin_amdgcn_tensor_load_to_lds)
#define HAVE_TDM 1
// TDM 2D tile load: tile_d1 rows x tile_d0 cols (4-byte elems) from a
// row-major tensor (line stride stride_d0 elems) into packed LDS at lds_off.
// D# layout per CDNA5 ISA.
__device__ __forceinline__ void tdm_load_2d(unsigned lds_off, const void* gaddr,
                                            unsigned tile_d0, unsigned tile_d1,
                                            unsigned tensor_d0, unsigned tensor_d1,
                                            unsigned stride_d0) {
  const unsigned long long ga = (unsigned long long)gaddr;
  u32x4 g0 = {1u,                                   // count=1, user mode
              lds_off,                              // lds_addr (bytes)
              (unsigned)(ga & 0xFFFFFFFFu),         // global_addr[31:0]
              (unsigned)(((ga >> 32) & 0x01FFFFFFu) | (2u << 30))}; // [56:32]|type=2
  i32x8 g1 = {(int)(2u << 16),                                      // data_size=4B
              (int)((tensor_d0 & 0xFFFFu) << 16),                   // dim0[15:0]
              (int)((tensor_d0 >> 16) | ((tensor_d1 & 0xFFFFu) << 16)),
              (int)((tensor_d1 >> 16) | (tile_d0 << 16)),           // tile_dim0
              (int)(tile_d1 & 0xFFFFu),                             // tile_dim1
              (int)stride_d0,                                       // dim0 stride
              0, 0};
  i32x4 z4 = {0, 0, 0, 0};
#if __clang_major__ >= 23
  i32x8 z8 = {0, 0, 0, 0, 0, 0, 0, 0};
  __builtin_amdgcn_tensor_load_to_lds(g0, g1, z4, z4, z8, 0);
#else
  __builtin_amdgcn_tensor_load_to_lds(g0, g1, z4, z4, 0);
#endif
}
#endif

// A fragment (16x32, row-major A[m,k], leading dim ld): ISA 7.12.2 layout.
__device__ __forceinline__ v16h load_a_rm(const float* A, int ld, int m0, int k0) {
  int lane = threadIdx.x & 31;
  const float* row = A + (size_t)(m0 + (lane & 15)) * ld + k0 + ((lane >> 4) << 3);
  v16h r;
#pragma unroll
  for (int e = 0; e < 16; e += 2) {
    int kk = (e & 7) + ((e >> 3) << 4);
    float2 v = *(const float2*)(row + kk);
    r[e] = (_Float16)v.x;
    r[e + 1] = (_Float16)v.y;
  }
  return r;
}

// A fragment where A[m,k] is stored at addr = k*strideK + m.
__device__ __forceinline__ v16h load_a_cm(const float* A, int strideK, int m0,
                                          int k0) {
  int lane = threadIdx.x & 31;
  int m = m0 + (lane & 15);
  int kh = (lane >> 4) << 3;
  v16h r;
#pragma unroll
  for (int e = 0; e < 16; ++e) {
    int kk = (e & 7) + ((e >> 3) << 4) + kh;
    r[e] = (_Float16)A[(size_t)(k0 + kk) * strideK + m];
  }
  return r;
}

// B fragment (32x16, row-major B[k,n], leading dim ld): lane = n, element = k (table order).
__device__ __forceinline__ v16h load_b_rm(const float* B, int ld, int k0, int n0) {
  int lane = threadIdx.x & 31;
  int n = n0 + (lane & 15);
  int kh = (lane >> 4) << 3;
  v16h r;
#pragma unroll
  for (int e = 0; e < 16; ++e) {
    int kk = (e & 7) + ((e >> 3) << 4) + kh;
    r[e] = (_Float16)B[(size_t)(k0 + kk) * ld + n];
  }
  return r;
}

// B fragment where B[k,n] is at addr = n*strideN + k (strided along n).
__device__ __forceinline__ v16h load_b_cm(const float* B, int strideN, int k0,
                                          int n0) {
  int lane = threadIdx.x & 31;
  int n = n0 + (lane & 15);
  int kh = (lane >> 4) << 3;
  v16h r;
#pragma unroll
  for (int e = 0; e < 16; ++e) {
    int kk = (e & 7) + ((e >> 3) << 4) + kh;
    r[e] = (_Float16)B[(size_t)n * strideN + k0 + kk];
  }
  return r;
}

// ------------------- per-channel GEMM: Y[bc,M,N] = W[c,M,K] * X[bc,K,N] ----
// Block: 256 threads (8 waves), 64x64 output tile, K staged in 32-slabs.
// W-slab (64x32) via TDM tensor_load_to_lds, X-slab (32x64) via async B128.
__global__ void mcl_kernel(const float* __restrict__ X, const float* __restrict__ Wm,
                           float* __restrict__ Y, int M, int K) {
  __shared__ __align__(16) float Ws[64 * 32];  // [m][k], ld 32 (8 KB)
  __shared__ __align__(16) float Xs[32 * 64];  // [k][n], ld 64 (8 KB)

  const int nblk = blockIdx.x * 64;
  const int mblk = blockIdx.y * 64;
  const int bc = blockIdx.z;
  const int c = bc & (CD - 1);
  const float* Wc = Wm + (size_t)c * M * K;
  const float* Xc = X + (size_t)bc * K * WD;
  float* Yc = Y + (size_t)bc * M * WD;

  const int t = threadIdx.x;
  const int wave = t >> 5;
  const int lane = t & 31;
  const int msub = wave >> 1;          // 0..3  -> m-tile
  const int nt0 = (wave & 1) << 1;     // 0 or 2 -> first of two n-tiles

  const unsigned wsBase = (unsigned)(size_t)&Ws[0];
  const unsigned xsBase = (unsigned)(size_t)&Xs[0];

  v8f acc0 = {0.f, 0.f, 0.f, 0.f, 0.f, 0.f, 0.f, 0.f};
  v8f acc1 = {0.f, 0.f, 0.f, 0.f, 0.f, 0.f, 0.f, 0.f};

  for (int k0 = 0; k0 < K; k0 += 32) {
#ifdef HAVE_TDM
    // W-slab: one TDM descriptor moves the 64x32 tile into packed LDS.
    if (wave == 0)
      tdm_load_2d(wsBase, Wc + (size_t)mblk * K + k0,
                  /*tile_d0=*/32u, /*tile_d1=*/64u,
                  /*tensor_d0=*/(unsigned)K, /*tensor_d1=*/(unsigned)M,
                  /*stride_d0=*/(unsigned)K);
#else
#pragma unroll
    for (int s = 0; s < 2; ++s) {
      const int ch = t + (s << 8);
      const int wm = ch >> 3, wk = (ch & 7) << 2;
      async_g2l_b128(wsBase + (unsigned)(wm * 32 + wk) * 4,
                     Wc + (size_t)(mblk + wm) * K + k0 + wk);
    }
#endif
    // X-slab: per-lane async copies (512 x 4-float chunks).
#pragma unroll
    for (int s = 0; s < 2; ++s) {
      const int ch = t + (s << 8);
      const int xk = ch >> 4, xn = (ch & 15) << 2;
      async_g2l_b128(xsBase + (unsigned)(xk * 64 + xn) * 4,
                     Xc + (size_t)(k0 + xk) * WD + nblk + xn);
    }
    if (k0 + 32 < K)
      __builtin_prefetch(Xc + (size_t)(k0 + 32 + lane) * WD + nblk, 0, 0);
    wait_async0();
#ifdef HAVE_TDM
    if (wave == 0) __builtin_amdgcn_s_wait_tensorcnt(0);
#endif
    __syncthreads();

    v16h a = load_a_rm(Ws, 32, msub * 16, 0);
    v16h b0 = load_b_rm(Xs, 64, 0, nt0 * 16);
    v16h b1 = load_b_rm(Xs, 64, 0, nt0 * 16 + 16);
    acc0 = wmma_f16(a, b0, acc0);
    acc1 = wmma_f16(a, b1, acc1);
    __syncthreads();
  }

  const int n = nblk + nt0 * 16 + (lane & 15);
  const int mb = mblk + msub * 16 + ((lane >> 4) << 3);
#pragma unroll
  for (int v = 0; v < 8; ++v) {
    Yc[(size_t)(mb + v) * WD + n] = acc0[v];
    Yc[(size_t)(mb + v) * WD + n + 16] = acc1[v];
  }
}

// ------------------- fused attention core ----------------------------------
// Q,K,V in [bc, f=h*64+d, w] layout. One wave per (bc, head, q-tile of 16).
// qk_out[b,c,h,q,k] = prev[...] + (Q.K)/8 ; P = softmax_k(qk) ; A = P*V.
__global__ void attn_core_kernel(const float* __restrict__ Qb,
                                 const float* __restrict__ Kb,
                                 const float* __restrict__ Vb,
                                 const float* __restrict__ prev,
                                 float* __restrict__ qk_out,
                                 float* __restrict__ Aout) {
  __shared__ float P[16][WD];  // 16 KB
  const int qt = blockIdx.x, hh = blockIdx.y, bc = blockIdx.z;
  const int lane = threadIdx.x;
  const float* Q  = Qb + ((size_t)bc * FD + hh * HD_) * WD;
  const float* Km = Kb + ((size_t)bc * FD + hh * HD_) * WD;
  const float* Vm = Vb + ((size_t)bc * FD + hh * HD_) * WD;
  const size_t qkbase = (((size_t)bc * HEADS_ + hh) * WD) * WD;
  const int q0 = qt * 16;
  const float scale = 0.125f;  // 1/sqrt(64)

  v16h aq0 = load_a_cm(Q, WD, q0, 0);
  v16h aq1 = load_a_cm(Q, WD, q0, 32);

  for (int nt = 0; nt < 16; ++nt) {
    v8f acc = {0.f, 0.f, 0.f, 0.f, 0.f, 0.f, 0.f, 0.f};
    v16h b0 = load_b_rm(Km, WD, 0, nt * 16);
    v16h b1 = load_b_rm(Km, WD, 32, nt * 16);
    acc = wmma_f16(aq0, b0, acc);
    acc = wmma_f16(aq1, b1, acc);
    const int n = nt * 16 + (lane & 15);
    const int mb = (lane >> 4) << 3;
#pragma unroll
    for (int v = 0; v < 8; ++v) {
      const int m = mb + v;
      const float val = prev[qkbase + (size_t)(q0 + m) * WD + n] + acc[v] * scale;
      qk_out[qkbase + (size_t)(q0 + m) * WD + n] = val;
      P[m][n] = val;
    }
  }
  __syncthreads();

  for (int r = 0; r < 16; ++r) {
    float mx = -1e30f;
    for (int c0 = lane; c0 < WD; c0 += 32) mx = fmaxf(mx, P[r][c0]);
#pragma unroll
    for (int off = 16; off > 0; off >>= 1) mx = fmaxf(mx, __shfl_xor(mx, off, 32));
    float sum = 0.f;
    for (int c0 = lane; c0 < WD; c0 += 32) {
      float e = __expf(P[r][c0] - mx);
      P[r][c0] = e;
      sum += e;
    }
#pragma unroll
    for (int off = 16; off > 0; off >>= 1) sum += __shfl_xor(sum, off, 32);
    const float inv = 1.0f / sum;
    for (int c0 = lane; c0 < WD; c0 += 32) P[r][c0] *= inv;
  }
  __syncthreads();

#pragma unroll
  for (int dt = 0; dt < 4; ++dt) {
    v8f acc = {0.f, 0.f, 0.f, 0.f, 0.f, 0.f, 0.f, 0.f};
    for (int ks = 0; ks < 8; ++ks) {
      v16h a = load_a_rm(&P[0][0], WD, 0, ks * 32);
      v16h b = load_b_cm(Vm, WD, ks * 32, dt * 16);
      acc = wmma_f16(a, b, acc);
    }
    const int d = dt * 16 + (lane & 15);
    const int mb = (lane >> 4) << 3;
#pragma unroll
    for (int v = 0; v < 8; ++v) {
      const int q = q0 + mb + v;
      Aout[((size_t)bc * FD + hh * HD_ + d) * WD + q] = acc[v];
    }
  }
}

// ------------------------- VALU kernels ------------------------------------

__global__ void embed_kernel(const float* __restrict__ x, const float* __restrict__ Wv,
                             const float* __restrict__ bv, float* __restrict__ h) {
  size_t i = (size_t)blockIdx.x * blockDim.x + threadIdx.x;
  if (i >= FSZ) return;
  const size_t fw = i % ((size_t)FD * WD);
  const int o = (int)((i / ((size_t)FD * WD)) % CD);
  const int b = (int)(i / ((size_t)CD * FD * WD));
  float acc = bv[o];
  const float* xb = x + (size_t)b * 16 * FD * WD + fw;
#pragma unroll
  for (int c = 0; c < 16; ++c) acc += Wv[o * 16 + c] * xb[(size_t)c * FD * WD];
  h[i] = acc;
}

__global__ void ln_kernel(const float* __restrict__ X, float* __restrict__ Y,
                          const float* __restrict__ gamma,
                          const float* __restrict__ beta) {
  __shared__ float red[256];
  const int bc = blockIdx.x >> 8;
  const int w = blockIdx.x & 255;
  const int c = bc & (CD - 1);
  const int t = threadIdx.x;
  const float* Xb = X + (size_t)bc * FD * WD + w;
  const float v0 = Xb[(size_t)t * WD];
  const float v1 = Xb[(size_t)(t + 256) * WD];
  red[t] = v0 + v1;
  __syncthreads();
  for (int o = 128; o > 0; o >>= 1) {
    if (t < o) red[t] += red[t + o];
    __syncthreads();
  }
  const float mean = red[0] * (1.0f / FD);
  __syncthreads();
  const float d0 = v0 - mean, d1 = v1 - mean;
  red[t] = d0 * d0 + d1 * d1;
  __syncthreads();
  for (int o = 128; o > 0; o >>= 1) {
    if (t < o) red[t] += red[t + o];
    __syncthreads();
  }
  const float rstd = rsqrtf(red[0] * (1.0f / FD) + 1e-5f);
  float* Yb = Y + (size_t)bc * FD * WD + w;
  Yb[(size_t)t * WD] = d0 * rstd * gamma[c * FD + t] + beta[c * FD + t];
  Yb[(size_t)(t + 256) * WD] = d1 * rstd * gamma[c * FD + t + 256] + beta[c * FD + t + 256];
}

__global__ void convtime_kernel(const float* __restrict__ X,
                                const float* __restrict__ Wc, float* __restrict__ Y) {
  size_t i = (size_t)blockIdx.x * blockDim.x + threadIdx.x;
  if (i >= FSZ) return;
  const int w = (int)(i & 255);
  const int f = (int)((i >> 8) & 511);
  const int o = (int)((i >> 17) & 7);
  const int b = (int)(i >> 20);
  float acc = 0.f;
#pragma unroll
  for (int c = 0; c < CD; ++c) {
    const float* xr = X + (((size_t)(b * CD + c) * FD) + f) * WD + w;
    const float* wr = Wc + (o * CD + c) * 3;
    if (w > 0) acc += wr[0] * xr[-1];
    acc += wr[1] * xr[0];
    if (w < WD - 1) acc += wr[2] * xr[1];
  }
  Y[i] = acc;
}

__global__ void rope_kernel(float* __restrict__ X) {
  size_t i = (size_t)blockIdx.x * blockDim.x + threadIdx.x;
  if (i >= (size_t)BD * CD * HEADS_ * 32 * WD) return;
  const int w = (int)(i & 255);
  const int p = (int)((i >> 8) & 31);
  const int hh = (int)((i >> 13) & 7);
  const int bc = (int)(i >> 16);
  const float inv = __powf(10000.0f, -(float)p / 32.0f);
  float s, c;
  __sincosf((float)w * inv, &s, &c);
  const size_t base = ((size_t)bc * FD + hh * HD_ + 2 * p) * WD + w;
  const float xe = X[base], xo = X[base + WD];
  X[base] = xe * c - xo * s;
  X[base + WD] = xe * s + xo * c;
}

__global__ void frameconv_kernel(const float* __restrict__ X,
                                 const float* __restrict__ Wc, float* __restrict__ Y,
                                 int k, int do_sqrelu) {
  size_t i = (size_t)blockIdx.x * blockDim.x + threadIdx.x;
  if (i >= FSZ) return;
  const int w = (int)(i & 255);
  const int f = (int)((i >> 8) & 511);
  const int c = (int)((i >> 17) & 7);
  const int b = (int)(i >> 20);
  const int p = (k - 1) >> 1;
  const size_t base = ((size_t)(b * CD + c) * FD) * WD + w;
  float acc = 0.f;
  for (int t = 0; t < k; ++t) {
    const int ff = f + t - p;
    if (ff >= 0 && ff < FD) acc += Wc[c * k + t] * X[base + (size_t)ff * WD];
  }
  if (do_sqrelu) {
    acc = fmaxf(acc, 0.f);
    acc *= acc;
  }
  Y[i] = acc;
}

__global__ void gate_conv3d_kernel(const float* __restrict__ prevq,
                                   const float* __restrict__ skipq,
                                   const float* __restrict__ W1,
                                   const float* __restrict__ b1,
                                   float* __restrict__ Z) {
  size_t i = (size_t)blockIdx.x * blockDim.x + threadIdx.x;
  if (i >= QKSZ) return;
  const int x = (int)(i & 255);
  const int y = (int)((i >> 8) & 255);
  const int d = (int)((i >> 16) & 7);
  const int o = (int)((i >> 19) & 7);
  const int b = (int)(i >> 22);
  float acc = b1[o];
  for (int c = 0; c < 16; ++c) {
    const float* src = (c < 8)
        ? prevq + (size_t)(b * CD + c) * HEADS_ * WD * WD
        : skipq + (size_t)(b * CD + (c - 8)) * HEADS_ * WD * WD;
    const float* wk = W1 + (size_t)(o * 16 + c) * 27;
#pragma unroll
    for (int kd = 0; kd < 3; ++kd) {
      const int dd = d + kd - 1;
      if (dd < 0 || dd > 7) continue;
#pragma unroll
      for (int ky = 0; ky < 3; ++ky) {
        const int yy = y + ky - 1;
        if (yy < 0 || yy > 255) continue;
        const float* row = src + ((size_t)dd * WD + yy) * WD;
        const float* wr = wk + (kd * 3 + ky) * 3;
        if (x > 0) acc += wr[0] * row[x - 1];
        acc += wr[1] * row[x];
        if (x < 255) acc += wr[2] * row[x + 1];
      }
    }
  }
  const float r = fmaxf(acc, 0.f);
  Z[i] = r * r;
}

__global__ void gate_g_kernel(const float* __restrict__ Z, const float* __restrict__ w2,
                              const float* __restrict__ b2, float* __restrict__ G) {
  size_t i = (size_t)blockIdx.x * blockDim.x + threadIdx.x;
  if (i >= GSZ) return;
  const size_t sp = i & (((size_t)HEADS_ * WD * WD) - 1);
  const int b = (int)(i >> 19);
  float acc = b2[0];
#pragma unroll
  for (int o = 0; o < CD; ++o)
    acc += w2[o] * Z[(size_t)(b * CD + o) * HEADS_ * WD * WD + sp];
  G[i] = 1.0f / (1.0f + __expf(-acc));
}

__global__ void blend_kernel(const float* __restrict__ G, const float* __restrict__ prevq,
                             const float* __restrict__ skipq, float* __restrict__ out) {
  size_t i = (size_t)blockIdx.x * blockDim.x + threadIdx.x;
  if (i >= QKSZ) return;
  const size_t sp = i & (((size_t)HEADS_ * WD * WD) - 1);
  const int b = (int)(i >> 22);
  const float g = G[(size_t)b * HEADS_ * WD * WD + sp];
  out[i] = g * prevq[i] + (1.0f - g) * skipq[i];
}

__global__ void add_inplace_kernel(float* __restrict__ y, const float* __restrict__ x,
                                   size_t n) {
  size_t i = (size_t)blockIdx.x * blockDim.x + threadIdx.x;
  if (i < n) y[i] += x[i];
}

__global__ void sqrelu_inplace_kernel(float* __restrict__ y, size_t n) {
  size_t i = (size_t)blockIdx.x * blockDim.x + threadIdx.x;
  if (i < n) {
    const float r = fmaxf(y[i], 0.f);
    y[i] = r * r;
  }
}

__global__ void concat_out_kernel(const float* __restrict__ x, const float* __restrict__ h,
                                  float* __restrict__ out) {
  size_t i = (size_t)blockIdx.x * blockDim.x + threadIdx.x;
  if (i >= CATSZ) return;
  const size_t fw = i % ((size_t)FD * WD);
  const int c = (int)((i / ((size_t)FD * WD)) % 24);
  const int b = (int)(i / ((size_t)24 * FD * WD));
  out[i] = (c < 16) ? x[(size_t)(b * 16 + c) * FD * WD + fw]
                    : h[(size_t)(b * CD + (c - 16)) * FD * WD + fw];
}

// ------------------------------ host side ----------------------------------

enum {
  IN_X = 0, IN_SKIP, IN_PQK1, IN_PQK2, IN_SKIPQK,
  IN_EMB_W, IN_EMB_B,
  IN_G1_W1, IN_G1_B1, IN_G1_W2, IN_G1_B2,
  IN_G2_W1, IN_G2_B1, IN_G2_W2, IN_G2_B2,
  IN_NORM = 15,           // 15..26 : g1,b1,...,g6,b6
  IN_A1A = 27, IN_A1B = 34, IN_A2 = 41, IN_A3 = 48,
  IN_C1A = 55, IN_C1B = 56, IN_C2 = 57, IN_C3 = 58, IN_C4 = 59
};

extern "C" void kernel_launch(void* const* d_in, const int* in_sizes, int n_in,
                              void* d_out, int out_size, void* d_ws, size_t ws_size,
                              hipStream_t stream) {
  const size_t need = (7 * FSZ + 3 * QKSZ + GSZ) * sizeof(float);
  if (ws_size < need || n_in < 60) return;

  auto in = [&](int i) { return (const float*)d_in[i]; };
  const float* x = in(IN_X);
  const float* skip = in(IN_SKIP);
  const float* pqk1_in = in(IN_PQK1);
  const float* pqk2_in = in(IN_PQK2);
  const float* skipqk = in(IN_SKIPQK);

  float* ws = (float*)d_ws;
  float* h   = ws;
  float* z   = h + FSZ;
  float* t1  = z + FSZ;
  float* t2  = t1 + FSZ;
  float* qb  = t2 + FSZ;
  float* kb  = qb + FSZ;
  float* vb  = kb + FSZ;
  float* qkA = vb + FSZ;   // QKSZ
  float* qkB = qkA + QKSZ; // QKSZ
  float* big = qkB + QKSZ; // QKSZ (gate z1 / MLP mid)
  float* gg  = big + QKSZ; // GSZ

  float* out_cat = (float*)d_out;
  float* out_qk1 = out_cat + CATSZ;
  float* out_qk2 = out_qk1 + QKSZ;

  const int TB = 256;
  auto blocks = [&](size_t n) { return (unsigned)((n + TB - 1) / TB); };

  auto mcl = [&](const float* X, const float* Wm, float* Y, int M, int K) {
    mcl_kernel<<<dim3(WD / 64, M / 64, BD * CD), 256, 0, stream>>>(X, Wm, Y, M, K);
  };
  auto LN = [&](const float* X, float* Y, int idx) {
    ln_kernel<<<BD * CD * WD, 256, 0, stream>>>(X, Y, in(IN_NORM + 2 * idx),
                                                in(IN_NORM + 2 * idx + 1));
  };
  auto addh = [&](const float* src) {
    add_inplace_kernel<<<blocks(FSZ), TB, 0, stream>>>(h, src, FSZ);
  };
  auto run_attn = [&](const float* xin, const float* src, const float* prev,
                      float* qkout, int pb) {
    mcl(xin, in(pb + 0), t1, FD, FD);                                     // wq
    convtime_kernel<<<blocks(FSZ), TB, 0, stream>>>(t1, in(pb + 1), qb);  // cq
    rope_kernel<<<blocks(FSZ / 2), TB, 0, stream>>>(qb);
    mcl(src, in(pb + 2), t1, FD, FD);                                     // wk
    convtime_kernel<<<blocks(FSZ), TB, 0, stream>>>(t1, in(pb + 3), kb);  // ck
    rope_kernel<<<blocks(FSZ / 2), TB, 0, stream>>>(kb);
    mcl(src, in(pb + 4), t1, FD, FD);                                     // wv
    convtime_kernel<<<blocks(FSZ), TB, 0, stream>>>(t1, in(pb + 5), vb);  // cv
    attn_core_kernel<<<dim3(16, HEADS_, BD * CD), 32, 0, stream>>>(
        qb, kb, vb, prev, qkout, t1);
    mcl(t1, in(pb + 6), t2, FD, FD);                                      // wo
    addh(t2);
  };
  auto run_gate = [&](const float* prevsrc, float* dst, int w1, int b1, int w2,
                      int b2) {
    gate_conv3d_kernel<<<blocks(QKSZ), TB, 0, stream>>>(prevsrc, skipqk, in(w1),
                                                        in(b1), big);
    gate_g_kernel<<<blocks(GSZ), TB, 0, stream>>>(big, in(w2), in(b2), gg);
    blend_kernel<<<blocks(QKSZ), TB, 0, stream>>>(gg, prevsrc, skipqk, dst);
  };

  // ---- embed + block 1 ----
  embed_kernel<<<blocks(FSZ), TB, 0, stream>>>(x, in(IN_EMB_W), in(IN_EMB_B), h);
  LN(h, z, 0);                                    // norm1
  run_attn(z, z, pqk1_in, qkA, IN_A1A);           // attn1a, qkA = new prev_qk1
  run_gate(pqk2_in, qkB, IN_G1_W1, IN_G1_B1, IN_G1_W2, IN_G1_B2);  // qkB = blended
  run_attn(z, skip, qkB, qkB, IN_A1B);            // attn1b (in-place qk update)

  // ---- conv block ----
  LN(h, z, 1);                                                       // norm2
  frameconv_kernel<<<blocks(FSZ), TB, 0, stream>>>(z, in(IN_C1A), t1, 11, 1);
  frameconv_kernel<<<blocks(FSZ), TB, 0, stream>>>(z, in(IN_C1B), t2, 7, 0);
  add_inplace_kernel<<<blocks(FSZ), TB, 0, stream>>>(t1, t2, FSZ);
  LN(t1, t2, 2);                                                     // norm3
  frameconv_kernel<<<blocks(FSZ), TB, 0, stream>>>(t2, in(IN_C2), t1, 7, 0);
  addh(t1);

  // ---- attn2 (self) : final prev_qk1 straight to d_out ----
  LN(h, z, 3);                                    // norm4
  run_attn(z, z, qkA, out_qk1, IN_A2);

  // ---- gate2 + attn3 (cross) : final prev_qk2 straight to d_out ----
  run_gate(qkB, qkA, IN_G2_W1, IN_G2_B1, IN_G2_W2, IN_G2_B2);  // qkA = blended
  LN(h, z, 4);                                    // norm5
  run_attn(z, skip, qkA, out_qk2, IN_A3);

  // ---- MLP: 512 -> 2048 -> sq_relu -> 512 ----
  LN(h, z, 5);                                    // norm6
  mcl(z, in(IN_C3), big, EXP_, FD);
  sqrelu_inplace_kernel<<<blocks(QKSZ), TB, 0, stream>>>(big, QKSZ);
  mcl(big, in(IN_C4), t1, FD, EXP_);
  addh(t1);

  // ---- outputs ----
  concat_out_kernel<<<blocks(CATSZ), TB, 0, stream>>>(x, h, out_cat);
}
